// GPTModel_57767310131668
// MI455X (gfx1250) — compile-verified, matrix operand order checked by reading, not run on hardware
//
#include <hip/hip_runtime.h>
#include <cstddef>
#include <cstdint>

// ---------------------------------------------------------------------------
// MI455X (gfx1250, wave32) GPT forward: embed -> QKV -> causal scores ->
// softmax(axis=query) -> att@v -> lm_head(+bias) -> log_softmax NLL.
// GEMMs: v_wmma_f32_16x16x32_bf16, double-buffered LDS B tiles staged with
// GLOBAL_LOAD_ASYNC_TO_LDS_B128 (ASYNCcnt) when available.
// ---------------------------------------------------------------------------

typedef __attribute__((ext_vector_type(16))) __bf16 v16bf;
typedef __attribute__((ext_vector_type(8)))  __bf16 v8bf;
typedef __attribute__((ext_vector_type(8)))  float  v8f;
typedef __attribute__((ext_vector_type(4)))  int    v4i;

#define EMB   512
#define TLEN  1024
#define BATCH 4
#define ROWS  (BATCH * TLEN)      // 4096
#define VOCAB 50257
#define VPAD  50304               // 128 * 393
#define LROW  72                  // padded LDS row (bf16 elems): 144B, 16B-aligned

#if defined(__has_builtin)
#  if __has_builtin(__builtin_amdgcn_global_load_async_to_lds_b128)
#    define USE_ASYNC_LDS 1
#  endif
#endif
#ifndef USE_ASYNC_LDS
#  define USE_ASYNC_LDS 0
#endif

struct StageRegs { v8bf r0, r1, r2, r3; };

// Stage 32 bf16 (4x16B) from a global row into LDS. issue = start loads,
// commit = make them visible in LDS (async: wait ASYNCcnt; sync: ds_store).
__device__ __forceinline__
void stage_issue(const __bf16* g, __bf16* l, StageRegs& s)
{
#if USE_ASYNC_LDS
    auto gp = (__attribute__((address_space(1))) v4i*)(const_cast<__bf16*>(g));
    auto lp = (__attribute__((address_space(3))) v4i*)(l);
    __builtin_amdgcn_global_load_async_to_lds_b128(gp, lp, 0,  0);
    __builtin_amdgcn_global_load_async_to_lds_b128(gp, lp, 16, 0);
    __builtin_amdgcn_global_load_async_to_lds_b128(gp, lp, 32, 0);
    __builtin_amdgcn_global_load_async_to_lds_b128(gp, lp, 48, 0);
    (void)s;
#else
    s.r0 = *(const v8bf*)(g);
    s.r1 = *(const v8bf*)(g + 8);
    s.r2 = *(const v8bf*)(g + 16);
    s.r3 = *(const v8bf*)(g + 24);
    (void)l;
#endif
}

__device__ __forceinline__
void stage_commit(__bf16* l, StageRegs& s)
{
#if USE_ASYNC_LDS
#  if __has_builtin(__builtin_amdgcn_s_wait_asynccnt)
    __builtin_amdgcn_s_wait_asynccnt(0);
#  else
    asm volatile("s_wait_asynccnt 0x0" ::: "memory");
#  endif
    (void)l; (void)s;
#else
    *(v8bf*)(l)      = s.r0;
    *(v8bf*)(l + 8)  = s.r1;
    *(v8bf*)(l + 16) = s.r2;
    *(v8bf*)(l + 24) = s.r3;
#endif
}

// ---------------------------------------------------------------------------
// Weight transpose + fp32->bf16 convert:  W[K][N] -> Wt[Npad][K] (pad rows = 0)
// ---------------------------------------------------------------------------
__global__ __launch_bounds__(256)
void transpose_to_bf16_kernel(const float* __restrict__ W, __bf16* __restrict__ Wt,
                              int K, int N)
{
    __shared__ float tile[32][33];
    const int tx = threadIdx.x & 31;
    const int ty = threadIdx.x >> 5;          // 0..7
    const int n  = blockIdx.x * 32 + tx;
    const int kb = blockIdx.y * 32;
    for (int i = 0; i < 4; ++i) {
        const int k = kb + ty + i * 8;
        tile[ty + i * 8][tx] = (n < N) ? W[(size_t)k * N + n] : 0.0f;
    }
    __syncthreads();
    for (int i = 0; i < 4; ++i) {
        const int np = blockIdx.x * 32 + ty + i * 8;
        Wt[(size_t)np * K + kb + tx] = (__bf16)tile[tx][ty + i * 8];
    }
}

// ---------------------------------------------------------------------------
// x = tok_emb[idx] + pos_emb[t]  -> bf16 [4096][512]
// ---------------------------------------------------------------------------
__global__ __launch_bounds__(256)
void embed_kernel(const int* __restrict__ idx,
                  const float* __restrict__ tok, const float* __restrict__ pos,
                  __bf16* __restrict__ xout)
{
    const int row = blockIdx.x;               // 0..4095
    const int t   = row & (TLEN - 1);
    const int token = idx[row];
    const float* te = tok + (size_t)token * EMB;
    const float* pe = pos + (size_t)t * EMB;
    __bf16* xr = xout + (size_t)row * EMB;
    for (int e = threadIdx.x; e < EMB; e += 256)
        xr[e] = (__bf16)(te[e] + pe[e]);
}

// ---------------------------------------------------------------------------
// bf16 WMMA GEMM: C[M][N] = scale * (A[M][K] @ Bt[N][K]^T)
// block = 256 threads = 8 waves -> C tile 256(M) x 128(N); wave w owns rows
// [by*256 + w*32, +32) as 2 M-subtiles x 8 N-subtiles = 16 accumulators.
// B tile (128 x 64K bf16) double-buffered through LDS; software-pipelined:
// issue next stage -> 32 WMMAs on current -> commit -> barrier.
// Requires: M % 256 == 0, N % 128 == 0, K % 64 == 0.
// mode 0: f32 store (+bias, col<n_limit)         -> Cf
// mode 1: f32 store, causal mask (col<=row keep) -> Cf
// mode 2: bf16 row-major store                   -> Cb
// mode 3: bf16 transposed store Cb[col*ldc+row]  -> Cb
// ---------------------------------------------------------------------------
__global__ __launch_bounds__(256)
void gemm_bf16_kernel(const __bf16* __restrict__ A, int lda,
                      const __bf16* __restrict__ Bt, int ldb,
                      float* __restrict__ Cf, __bf16* __restrict__ Cb, int ldc,
                      int M, int N, int K,
                      float scale, const float* __restrict__ bias,
                      int n_limit, int mode)
{
    __shared__ __bf16 btile[2][128 * LROW];

    const int tid     = threadIdx.x;
    const int lane    = tid & 31;
    const int wave    = tid >> 5;
    const int m0      = blockIdx.y * 256 + wave * 32;
    const int nblk    = blockIdx.x * 128;
    const int rowsel  = lane & 15;
    const int halfsel = lane >> 4;            // K-octet select (0 or 1)

    // staging map: thread -> (B row, 32-elem K chunk of the 64-wide slab)
    const int st_row = tid >> 1;              // 0..127
    const int st_k   = (tid & 1) * 32;        // 0 or 32
    const __bf16* gstage = Bt + (size_t)(nblk + st_row) * ldb + st_k;
    __bf16* lstage[2] = { &btile[0][st_row * LROW + st_k],
                          &btile[1][st_row * LROW + st_k] };

    v8f acc[2][8];
    #pragma unroll
    for (int mi = 0; mi < 2; ++mi)
        #pragma unroll
        for (int nj = 0; nj < 8; ++nj) acc[mi][nj] = (v8f){};

    StageRegs sreg;
    stage_issue(gstage, lstage[0], sreg);
    stage_commit(lstage[0], sreg);
    __syncthreads();

    const int nsteps = K >> 6;                // K / 64
    for (int i = 0; i < nsteps; ++i) {
        const int k0  = i << 6;
        const int cur = i & 1;
        const bool has_next = (i + 1) < nsteps;
        if (has_next) stage_issue(gstage + k0 + 64, lstage[cur ^ 1], sreg);

        #pragma unroll
        for (int kk = 0; kk < 64; kk += 32) {
            // A fragments for the two M-subtiles (global, b128 pairs)
            v16bf af[2];
            #pragma unroll
            for (int mi = 0; mi < 2; ++mi) {
                const __bf16* ar = A + (size_t)(m0 + mi * 16 + rowsel) * lda
                                     + k0 + kk + halfsel * 8;
                v8bf lo = *(const v8bf*)(ar);
                v8bf hi = *(const v8bf*)(ar + 16);
                af[mi] = __builtin_shufflevector(lo, hi,
                    0,1,2,3,4,5,6,7,8,9,10,11,12,13,14,15);
            }
            // 8 B fragments from LDS, 16 WMMAs
            #pragma unroll
            for (int nj = 0; nj < 8; ++nj) {
                const __bf16* br = &btile[cur][(nj * 16 + rowsel) * LROW
                                              + kk + halfsel * 8];
                v8bf blo = *(const v8bf*)(br);
                v8bf bhi = *(const v8bf*)(br + 16);
                v16bf bf = __builtin_shufflevector(blo, bhi,
                    0,1,2,3,4,5,6,7,8,9,10,11,12,13,14,15);
                acc[0][nj] = __builtin_amdgcn_wmma_f32_16x16x32_bf16(
                    false, af[0], false, bf, (short)0, acc[0][nj], false, false);
                acc[1][nj] = __builtin_amdgcn_wmma_f32_16x16x32_bf16(
                    false, af[1], false, bf, (short)0, acc[1][nj], false, false);
            }
        }
        if (has_next) stage_commit(lstage[cur ^ 1], sreg);
        __syncthreads();
    }

    // Epilogue. C/D layout: VGPR r, lanes 0-15 -> row +r, lanes 16-31 -> +8+r.
    const float NEG_INF = -__builtin_inff();
    #pragma unroll
    for (int mi = 0; mi < 2; ++mi) {
        #pragma unroll
        for (int nj = 0; nj < 8; ++nj) {
            const int col = nblk + nj * 16 + rowsel;
            #pragma unroll
            for (int r = 0; r < 8; ++r) {
                const int orow = m0 + mi * 16 + r + halfsel * 8;
                float v = acc[mi][nj][r] * scale;
                if (mode == 0) {
                    if (col < n_limit) {
                        if (bias) v += bias[col];
                        Cf[(size_t)orow * ldc + col] = v;
                    }
                } else if (mode == 1) {
                    Cf[(size_t)orow * ldc + col] = (col <= orow) ? v : NEG_INF;
                } else if (mode == 2) {
                    Cb[(size_t)orow * ldc + col] = (__bf16)v;
                } else {
                    Cb[(size_t)col * ldc + orow] = (__bf16)v;
                }
            }
        }
    }
    (void)M; (void)N;
}

// ---------------------------------------------------------------------------
// Softmax over the QUERY axis (axis=1, faithful to reference):
// att[b,t,s] = exp(wei[b,t,s]-max_t) / sum_t.  Block = 64 cols x 4 t-stripes.
// ---------------------------------------------------------------------------
__global__ __launch_bounds__(256)
void colsoftmax_kernel(const float* __restrict__ wei, __bf16* __restrict__ att)
{
    __shared__ float sm[4][64], ss[4][64];
    const int b  = blockIdx.y;
    const int tx = threadIdx.x & 63;
    const int ty = threadIdx.x >> 6;          // 0..3
    const int s  = blockIdx.x * 64 + tx;
    const float* wb = wei + (size_t)b * TLEN * TLEN;
    __bf16*      ab = att + (size_t)b * TLEN * TLEN;
    const float NEG_INF = -__builtin_inff();

    float m = NEG_INF, sum = 0.f;
    for (int t = ty; t < TLEN; t += 4) {
        const float v = wb[t * TLEN + s];
        if (v == NEG_INF) continue;
        if (v > m) { sum = sum * __expf(m - v) + 1.0f; m = v; }
        else       { sum += __expf(v - m); }
    }
    sm[ty][tx] = m; ss[ty][tx] = sum;
    __syncthreads();
    if (ty == 0) {
        float M = sm[0][tx], S = ss[0][tx];
        #pragma unroll
        for (int i = 1; i < 4; ++i) {
            const float mi = sm[i][tx], si = ss[i][tx];
            const float Mn = fmaxf(M, mi);
            const float wa = (M  == NEG_INF) ? 0.f : __expf(M  - Mn);
            const float wbv= (mi == NEG_INF) ? 0.f : __expf(mi - Mn);
            S = S * wa + si * wbv; M = Mn;
        }
        sm[0][tx] = M; ss[0][tx] = (S > 0.f) ? (1.0f / S) : 0.f;
    }
    __syncthreads();
    const float Mf = sm[0][tx], Rs = ss[0][tx];
    for (int t = ty; t < TLEN; t += 4) {
        const float v = wb[t * TLEN + s];
        const float e = (v == NEG_INF) ? 0.f : __expf(v - Mf) * Rs;
        ab[t * TLEN + s] = (__bf16)e;
    }
}

// ---------------------------------------------------------------------------
// Per-row NLL: rowloss[row] = logsumexp(logits[row]) - logits[row][target].
// ---------------------------------------------------------------------------
__global__ __launch_bounds__(256)
void rowloss_kernel(const float* __restrict__ logits, const int* __restrict__ target,
                    float* __restrict__ rowloss)
{
    __shared__ float sm[256], ss[256];
    const int row = blockIdx.x;
    const float* lr = logits + (size_t)row * VOCAB;
    const float NEG_INF = -__builtin_inff();

    float m = NEG_INF, s = 0.f;
    for (int j = threadIdx.x; j < VOCAB; j += 256) {
        const float v = lr[j];
        if (v > m) { s = s * __expf(m - v) + 1.0f; m = v; }
        else       { s += __expf(v - m); }
    }
    sm[threadIdx.x] = m; ss[threadIdx.x] = s;
    __syncthreads();
    for (int off = 128; off > 0; off >>= 1) {
        if (threadIdx.x < off) {
            const float m1 = sm[threadIdx.x],       s1 = ss[threadIdx.x];
            const float m2 = sm[threadIdx.x + off], s2 = ss[threadIdx.x + off];
            const float Mn = fmaxf(m1, m2);
            const float a = (m1 == NEG_INF) ? 0.f : __expf(m1 - Mn);
            const float b = (m2 == NEG_INF) ? 0.f : __expf(m2 - Mn);
            sm[threadIdx.x] = Mn; ss[threadIdx.x] = s1 * a + s2 * b;
        }
        __syncthreads();
    }
    if (threadIdx.x == 0)
        rowloss[row] = (sm[0] + __logf(ss[0])) - lr[target[row]];
}

// Deterministic fixed-order final reduction of 4096 row losses -> mean.
__global__ __launch_bounds__(256)
void finalloss_kernel(const float* __restrict__ rowloss, float* __restrict__ out_loss)
{
    __shared__ float acc[256];
    float s = 0.f;
    for (int i = threadIdx.x; i < ROWS; i += 256) s += rowloss[i];
    acc[threadIdx.x] = s;
    __syncthreads();
    for (int off = 128; off > 0; off >>= 1) {
        if (threadIdx.x < off) acc[threadIdx.x] += acc[threadIdx.x + off];
        __syncthreads();
    }
    if (threadIdx.x == 0) *out_loss = acc[0] / (float)ROWS;
}

// ---------------------------------------------------------------------------
extern "C" void kernel_launch(void* const* d_in, const int* in_sizes, int n_in,
                              void* d_out, int out_size, void* d_ws, size_t ws_size,
                              hipStream_t stream)
{
    const int*   idx     = (const int*)  d_in[0];
    const int*   target  = (const int*)  d_in[1];
    const float* tok_emb = (const float*)d_in[2];
    const float* pos_emb = (const float*)d_in[3];
    const float* Wk      = (const float*)d_in[4];
    const float* Wq      = (const float*)d_in[5];
    const float* Wv      = (const float*)d_in[6];
    const float* W_lm    = (const float*)d_in[7];
    const float* b_lm    = (const float*)d_in[8];

    float* logits = (float*)d_out;                       // [4096][50257]
    float* loss   = logits + (size_t)ROWS * VOCAB;       // final slot

    // ---- workspace carve-up (256B aligned) ----
    char*  ws  = (char*)d_ws;
    size_t off = 0;
    auto carve = [&](size_t bytes) -> void* {
        void* p = ws + off;
        off = (off + bytes + 255) & ~(size_t)255;
        return p;
    };
    __bf16* xb    = (__bf16*)carve((size_t)ROWS * EMB * 2);
    __bf16* qb    = (__bf16*)carve((size_t)ROWS * EMB * 2);
    __bf16* kb    = (__bf16*)carve((size_t)ROWS * EMB * 2);
    __bf16* vt    = (__bf16*)carve((size_t)EMB * ROWS * 2);      // [h][b*1024+t]
    __bf16* outb  = (__bf16*)carve((size_t)ROWS * EMB * 2);
    __bf16* att   = (__bf16*)carve((size_t)BATCH * TLEN * TLEN * 2);
    float*  wei   = (float*) carve((size_t)BATCH * TLEN * TLEN * 4);
    __bf16* wqt   = (__bf16*)carve((size_t)EMB * EMB * 2);
    __bf16* wkt   = (__bf16*)carve((size_t)EMB * EMB * 2);
    __bf16* wvt   = (__bf16*)carve((size_t)EMB * EMB * 2);
    __bf16* wlmt  = (__bf16*)carve((size_t)VPAD * EMB * 2);
    float*  rloss = (float*) carve((size_t)ROWS * 4);
    (void)ws_size; (void)n_in; (void)in_sizes; (void)out_size;

    const float inv_sqrt_e = 0.044194173824159216f;      // 512^-0.5

    // 1) weights -> bf16 transposed [Npad][K]
    hipLaunchKernelGGL(transpose_to_bf16_kernel, dim3(EMB / 32, EMB / 32), dim3(256), 0, stream, Wq,   wqt,  EMB, EMB);
    hipLaunchKernelGGL(transpose_to_bf16_kernel, dim3(EMB / 32, EMB / 32), dim3(256), 0, stream, Wk,   wkt,  EMB, EMB);
    hipLaunchKernelGGL(transpose_to_bf16_kernel, dim3(EMB / 32, EMB / 32), dim3(256), 0, stream, Wv,   wvt,  EMB, EMB);
    hipLaunchKernelGGL(transpose_to_bf16_kernel, dim3(VPAD / 32, EMB / 32), dim3(256), 0, stream, W_lm, wlmt, EMB, VOCAB);

    // 2) embeddings
    hipLaunchKernelGGL(embed_kernel, dim3(ROWS), dim3(256), 0, stream, idx, tok_emb, pos_emb, xb);

    // 3) q, k (bf16) and v (bf16, transposed to [h][row])
    hipLaunchKernelGGL(gemm_bf16_kernel, dim3(EMB / 128, ROWS / 256), dim3(256), 0, stream,
                       xb, EMB, wqt, EMB, (float*)nullptr, qb, EMB,
                       ROWS, EMB, EMB, 1.0f, (const float*)nullptr, EMB, 2);
    hipLaunchKernelGGL(gemm_bf16_kernel, dim3(EMB / 128, ROWS / 256), dim3(256), 0, stream,
                       xb, EMB, wkt, EMB, (float*)nullptr, kb, EMB,
                       ROWS, EMB, EMB, 1.0f, (const float*)nullptr, EMB, 2);
    hipLaunchKernelGGL(gemm_bf16_kernel, dim3(EMB / 128, ROWS / 256), dim3(256), 0, stream,
                       xb, EMB, wvt, EMB, (float*)nullptr, vt, ROWS,
                       ROWS, EMB, EMB, 1.0f, (const float*)nullptr, EMB, 3);

    // 4) causal scores per batch: wei[t][s] = (q.k^T)*E^-0.5, s>t -> -inf
    for (int b = 0; b < BATCH; ++b) {
        hipLaunchKernelGGL(gemm_bf16_kernel, dim3(TLEN / 128, TLEN / 256), dim3(256), 0, stream,
                           qb + (size_t)b * TLEN * EMB, EMB,
                           kb + (size_t)b * TLEN * EMB, EMB,
                           wei + (size_t)b * TLEN * TLEN, (__bf16*)nullptr, TLEN,
                           TLEN, TLEN, EMB, inv_sqrt_e, (const float*)nullptr, TLEN, 1);
    }

    // 5) softmax over query axis -> att (bf16)
    hipLaunchKernelGGL(colsoftmax_kernel, dim3(TLEN / 64, BATCH), dim3(256), 0, stream, wei, att);

    // 6) out = att @ v  (B^T rows are v_t[h][b*1024 + s], row stride ROWS)
    for (int b = 0; b < BATCH; ++b) {
        hipLaunchKernelGGL(gemm_bf16_kernel, dim3(EMB / 128, TLEN / 256), dim3(256), 0, stream,
                           att + (size_t)b * TLEN * TLEN, TLEN,
                           vt + (size_t)b * TLEN, ROWS,
                           (float*)nullptr, outb + (size_t)b * TLEN * EMB, EMB,
                           TLEN, EMB, TLEN, 1.0f, (const float*)nullptr, EMB, 2);
    }

    // 7) logits = out @ W_lm + b_lm  (write f32 straight into d_out, clamp to VOCAB)
    hipLaunchKernelGGL(gemm_bf16_kernel, dim3(VPAD / 128, ROWS / 256), dim3(256), 0, stream,
                       outb, EMB, wlmt, EMB, logits, (__bf16*)nullptr, VOCAB,
                       ROWS, VPAD, EMB, 1.0f, b_lm, VOCAB, 0);

    // 8) loss = -mean(log_softmax(logits)[target])  (deterministic reduction)
    hipLaunchKernelGGL(rowloss_kernel, dim3(ROWS), dim3(256), 0, stream, logits, target, rloss);
    hipLaunchKernelGGL(finalloss_kernel, dim3(1), dim3(256), 0, stream, rloss, loss);
}
